// WindowGrapher_43439299232099
// MI455X (gfx1250) — matrix operand order repaired and checked by reading, not compiled, not run on hardware
//
#include <hip/hip_runtime.h>
#include <hip/hip_bf16.h>

// ---------------------------------------------------------------------------
// Types for gfx1250 (wave32) WMMA + TDM
// ---------------------------------------------------------------------------
typedef unsigned short u16;
typedef u16      u16x8   __attribute__((ext_vector_type(8)));
typedef u16      u16x16  __attribute__((ext_vector_type(16)));
typedef __bf16   bf16x16 __attribute__((ext_vector_type(16)));
typedef float    f32x8   __attribute__((ext_vector_type(8)));
typedef unsigned uintx4  __attribute__((ext_vector_type(4)));
typedef int      intx4   __attribute__((ext_vector_type(4)));
typedef int      intx8   __attribute__((ext_vector_type(8)));

#define NWIN  1024   // 16 batches * 8 * 8 windows
#define NODES 64     // 8x8 pixels per window
#define CIN   96
#define OEDGE 192
#define KNN   9

__device__ __forceinline__ u16 f2bf(float f) {
  unsigned u = __builtin_bit_cast(unsigned, f);
  u += 0x7FFFu + ((u >> 16) & 1u);          // round-to-nearest-even
  return (u16)(u >> 16);
}
__device__ __forceinline__ float bf2f(u16 b) {
  return __builtin_bit_cast(float, (unsigned)b << 16);
}

__device__ __forceinline__ f32x8 wmma_bf16(bf16x16 a, bf16x16 b, f32x8 c) {
  // (neg_a, A, neg_b, B, c_mod, C, reuse_a, reuse_b)
  return __builtin_amdgcn_wmma_f32_16x16x32_bf16(false, a, false, b, (short)0, c,
                                                 false, false);
}

// ---------------------------------------------------------------------------
// Tensor Data Mover: 1-row tile load Global -> LDS (bytes multiple of 8).
// D# per CDNA5 ISA ch.8: group0 = {count=1, lds_addr, global_addr[56:0], type=2}
// group1 = {data_size=8B, tensor_dim0=tile_dim0=n8, tensor_dim1=tile_dim1=1,
//           tensor_dim0_stride=n8}. Tracked by TENSORcnt.
// This toolchain exposes the 6-arg builtin:
//   (u32x4 g0, i32x8 g1, i32x4 g2, i32x4 g3, i32x8 extra, i32 cpol)
// ---------------------------------------------------------------------------
__device__ __forceinline__ void tdm_load_1d(void* lds_dst, const void* gsrc,
                                            unsigned bytes) {
  unsigned n8 = bytes >> 3;                              // 8-byte elements
  unsigned long long ga = (unsigned long long)gsrc;
  unsigned lds = (unsigned)(unsigned long long)lds_dst;  // LDS offset = low 32b
  uintx4 g0 = { 1u,                                      // count=1 (valid)
                lds,                                     // lds_addr
                (unsigned)ga,                            // global_addr[31:0]
                (unsigned)((ga >> 32) & 0x01FFFFFFu) | 0x80000000u }; // +type=2
  intx8 g1 = { (int)(3u << 16),                // data_size=8B, workgroup_mask=0
               (int)(n8 << 16),                // tensor_dim0[15:0]
               (int)((n8 >> 16) | (1u << 16)), // tensor_dim0[31:16], tensor_dim1=1
               (int)(n8 << 16),                // tile_dim0 = n8   (n8 < 65536)
               1,                              // tile_dim1=1, tile_dim2=0
               (int)n8,                        // tensor_dim0_stride[31:0]
               0, 0 };
  intx4 z4 = { 0, 0, 0, 0 };
  intx8 z8 = { 0, 0, 0, 0, 0, 0, 0, 0 };
  __builtin_amdgcn_tensor_load_to_lds(g0, g1, z4, z4, z8, 0);
}
__device__ __forceinline__ void tdm_wait_all() {
  __builtin_amdgcn_s_wait_tensorcnt(0);
}

// ---------------------------------------------------------------------------
// WMMA fragment loads (wave32 layouts per ISA 7.12.2)
// A fragment (16x32 bf16, row-major src [M][stride]):
// lane<16 holds row m, K = {kblk+0..7, kblk+16..23}; lane>=16: K = {+8..15,+24..31}
// ---------------------------------------------------------------------------
__device__ __forceinline__ bf16x16 load_a_frag(const u16* base, int stride,
                                               int mrow, int kblk, int half) {
  const u16* p0 = base + mrow * stride + kblk + (half ? 8 : 0);
  u16x8 lo = *(const u16x8*)p0;             // 16B aligned (stride mult of 8)
  u16x8 hi = *(const u16x8*)(p0 + 16);
  u16x16 r = __builtin_shufflevector(lo, hi, 0,1,2,3,4,5,6,7,8,9,10,11,12,13,14,15);
  return __builtin_bit_cast(bf16x16, r);
}
// B fragment (32x16 bf16): column N = lane%16; lane<16: K = {kblk+0..7, kblk+16..23}; lane>=16: K = {+8..15,+24..31}
__device__ __forceinline__ bf16x16 load_b_frag(const u16* base, int stride,
                                               int ncol, int kblk, int half) {
  const u16* p0 = base + ncol * stride + kblk + (half ? 8 : 0);
  u16x8 lo = *(const u16x8*)p0;
  u16x8 hi = *(const u16x8*)(p0 + 16);
  u16x16 r = __builtin_shufflevector(lo, hi, 0,1,2,3,4,5,6,7,8,9,10,11,12,13,14,15);
  return __builtin_bit_cast(bf16x16, r);
}

// NT independent accumulator chains interleaved -> no WMMA->WMMA NOP stalls.
template <int NT>
__device__ __forceinline__ void gemm_tiles(const u16* __restrict__ A, int sA,
                                           const u16* __restrict__ B, int sB,
                                           int kdim, const int* mts, const int* nts,
                                           int l16, int half, f32x8* acc) {
  for (int kb = 0; kb < kdim; kb += 32) {
#pragma unroll
    for (int j = 0; j < NT; j++) {
      bf16x16 af = load_a_frag(A, sA, mts[j] * 16 + l16, kb, half);
      bf16x16 bf = load_b_frag(B, sB, nts[j] * 16 + l16, kb, half);
      acc[j] = wmma_bf16(af, bf, acc[j]);
    }
  }
}

// ---------------------------------------------------------------------------
// Kernel 0: fold BN affines into bf16 weights + f32 biases
// ---------------------------------------------------------------------------
__global__ __launch_bounds__(256) void prep_kernel(
    const float* __restrict__ fc1_w, const float* __restrict__ fc1_b,
    const float* __restrict__ fc1_s, const float* __restrict__ fc1_t,
    const float* __restrict__ ew,    const float* __restrict__ eb,
    const float* __restrict__ es,    const float* __restrict__ et,
    const float* __restrict__ fc2_w, const float* __restrict__ fc2_b,
    const float* __restrict__ fc2_s, const float* __restrict__ fc2_t,
    u16* __restrict__ W1, float* __restrict__ b1,
    u16* __restrict__ Wp, u16* __restrict__ Wq, float* __restrict__ bpq,
    u16* __restrict__ W2, float* __restrict__ b2)
{
  int i = blockIdx.x * 256 + threadIdx.x;
  if (i < 9216) {                                        // 96*96
    int o = i / 96;
    W1[i] = f2bf(fc1_w[i] * fc1_s[o]);
  } else if (i < 27648) {                                // 192*96
    int j = i - 9216; int o = j / 96, c = j - o * 96;
    Wp[j] = f2bf((ew[o * 192 + c] - ew[o * 192 + 96 + c]) * es[o]);
  } else if (i < 46080) {
    int j = i - 27648; int o = j / 96, c = j - o * 96;
    Wq[j] = f2bf(ew[o * 192 + 96 + c] * es[o]);
  } else if (i < 64512) {                                // 96*192
    int j = i - 46080; int oc = j / 192;
    W2[j] = f2bf(fc2_w[j] * fc2_s[oc]);
  } else if (i < 64896) {
    int j = i - 64512;
    if (j < 96)        b1[j] = fc1_b[j] * fc1_s[j] + fc1_t[j];
    else if (j < 288) { int o = j - 96;  bpq[o] = eb[o] * es[o] + et[o]; }
    else              { int oc = j - 288; b2[oc] = fc2_b[oc] * fc2_s[oc] + fc2_t[oc]; }
  }
}

// ---------------------------------------------------------------------------
// Kernel 1: fc1 (1x1 conv + affine), per-window 64x96x96 WMMA GEMM.
// Weights staged via TDM while all threads transpose/convert x to bf16.
// ---------------------------------------------------------------------------
__global__ __launch_bounds__(256) void fc1_kernel(
    const float* __restrict__ x, const u16* __restrict__ W1,
    const float* __restrict__ b1, u16* __restrict__ nodes_bf)
{
  __shared__ __align__(16) u16 As[NODES * CIN];   // 12.3 KB
  __shared__ __align__(16) u16 Ws[CIN * CIN];     // 18.4 KB
  int w = blockIdx.x;
  int b = w >> 6, wy = (w >> 3) & 7, wx = w & 7;
  int tid = threadIdx.x;

  if (tid < 32) tdm_load_1d(Ws, W1, CIN * CIN * 2);     // async weight stage

  // Stage the 8x8 window, all 96 channels: (c,row) units of 8 contiguous floats
  for (int u = tid; u < 96 * 8; u += 256) {
    int c = u >> 3, r = u & 7;
    const float* gp = x + ((((size_t)b * 96 + c) * 64 + wy * 8 + r) * 64 + wx * 8);
    float4 f0 = *(const float4*)gp;
    float4 f1 = *(const float4*)(gp + 4);
    int nb = r * 8;
    As[(nb + 0) * CIN + c] = f2bf(f0.x);
    As[(nb + 1) * CIN + c] = f2bf(f0.y);
    As[(nb + 2) * CIN + c] = f2bf(f0.z);
    As[(nb + 3) * CIN + c] = f2bf(f0.w);
    As[(nb + 4) * CIN + c] = f2bf(f1.x);
    As[(nb + 5) * CIN + c] = f2bf(f1.y);
    As[(nb + 6) * CIN + c] = f2bf(f1.z);
    As[(nb + 7) * CIN + c] = f2bf(f1.w);
  }
  tdm_wait_all();
  __syncthreads();

  int wave = tid >> 5, lane = tid & 31, half = lane >> 4, l16 = lane & 15;
  int mts[3], nts[3];
  f32x8 acc[3] = {{}, {}, {}};
#pragma unroll
  for (int j = 0; j < 3; j++) {                 // tiles wave, wave+8, wave+16
    int t = wave + 8 * j; mts[j] = t & 3; nts[j] = t >> 2;
  }
  gemm_tiles<3>(As, CIN, Ws, CIN, 96, mts, nts, l16, half, acc);

  u16* outp = nodes_bf + (size_t)w * NODES * CIN;
#pragma unroll
  for (int j = 0; j < 3; j++) {
    int o = nts[j] * 16 + l16;
    float bias = b1[o];
#pragma unroll
    for (int r = 0; r < 8; r++) {
      int node = mts[j] * 16 + r + half * 8;    // D layout: M = r + 8*(lane>=16)
      outp[node * CIN + o] = f2bf(acc[j][r] + bias);
    }
  }
}

// ---------------------------------------------------------------------------
// Kernel 2: kNN per window. Normalize nodes, Gram matrix via WMMA,
// stable top-9 smallest-distance selection (ties -> lower index).
// ---------------------------------------------------------------------------
__global__ __launch_bounds__(256) void knn_kernel(
    const u16* __restrict__ nodes_bf, int* __restrict__ idx)
{
  __shared__ __align__(16) u16   Xs[NODES * CIN];     // 12.3 KB
  __shared__ __align__(16) float G[NODES * NODES];    // 16 KB
  __shared__ float rn[NODES];
  int w = blockIdx.x, tid = threadIdx.x;
  const u16* src = nodes_bf + (size_t)w * NODES * CIN;

  if (tid < 32) tdm_load_1d(Xs, src, NODES * CIN * 2);
  tdm_wait_all();
  __syncthreads();

  if (tid < 64) {
    float ss = 0.f;
    for (int c = 0; c < CIN; c++) { float v = bf2f(Xs[tid * CIN + c]); ss += v * v; }
    rn[tid] = 1.0f / fmaxf(sqrtf(ss), 1e-12f);
  }
  __syncthreads();
  for (int u = tid; u < NODES * CIN; u += 256)
    Xs[u] = f2bf(bf2f(Xs[u]) * rn[u / CIN]);
  __syncthreads();

  int wave = tid >> 5, lane = tid & 31, half = lane >> 4, l16 = lane & 15;
  int mts[2], nts[2];
  f32x8 acc[2] = {{}, {}};
#pragma unroll
  for (int j = 0; j < 2; j++) {                 // 4x4 tiles of 64x64 Gram
    int t = wave + 8 * j; mts[j] = t & 3; nts[j] = t >> 2;
  }
  gemm_tiles<2>(Xs, CIN, Xs, CIN, 96, mts, nts, l16, half, acc);
#pragma unroll
  for (int j = 0; j < 2; j++)
#pragma unroll
    for (int r = 0; r < 8; r++)
      G[(mts[j] * 16 + r + half * 8) * 64 + nts[j] * 16 + l16] = acc[j][r];
  __syncthreads();

  if (tid < 64) {
    float bd[KNN]; int bi[KNN];
#pragma unroll
    for (int k = 0; k < KNN; k++) { bd[k] = 3.0e38f; bi[k] = 0; }
    for (int m = 0; m < 64; m++) {
      float d = 2.0f - 2.0f * G[tid * 64 + m];  // unit vectors: |a-b|^2 = 2-2ab
      if (d < bd[KNN - 1]) {
        int k = KNN - 1;
        while (k > 0 && d < bd[k - 1]) { bd[k] = bd[k - 1]; bi[k] = bi[k - 1]; --k; }
        bd[k] = d; bi[k] = m;
      }
    }
    int* op = idx + ((size_t)w * NODES + tid) * KNN;
#pragma unroll
    for (int k = 0; k < KNN; k++) op[k] = bi[k];
  }
}

// ---------------------------------------------------------------------------
// Kernel 3: p = nodes*Wp^T + bpq ; q = nodes*Wq^T   (two 64x96x192 GEMMs)
// Three parallel TDM stages (Wp, Wq, node features) issued by waves 0/1/2.
// ---------------------------------------------------------------------------
__global__ __launch_bounds__(256) void pq_kernel(
    const u16* __restrict__ nodes_bf, const u16* __restrict__ Wp,
    const u16* __restrict__ Wq, const float* __restrict__ bpq,
    float* __restrict__ p, float* __restrict__ q)
{
  __shared__ __align__(16) u16 As[NODES * CIN];        // 12.3 KB
  __shared__ __align__(16) u16 Ws[2 * OEDGE * CIN];    // 73.7 KB (Wp | Wq)
  int w = blockIdx.x, tid = threadIdx.x;
  const u16* src = nodes_bf + (size_t)w * NODES * CIN;

  if (tid < 32)       tdm_load_1d(Ws,               Wp, OEDGE * CIN * 2);
  else if (tid < 64)  tdm_load_1d(Ws + OEDGE * CIN, Wq, OEDGE * CIN * 2);
  else if (tid < 96)  tdm_load_1d(As,               src, NODES * CIN * 2);
  tdm_wait_all();
  __syncthreads();

  int wave = tid >> 5, lane = tid & 31, half = lane >> 4, l16 = lane & 15;
  for (int g = 0; g < 4; g++) {               // 12 tiles/wave in groups of 3
    int sel = g >> 1;                         // uniform within group
    const u16* wsel = Ws + sel * OEDGE * CIN;
    int mts[3], nts[3];
    f32x8 acc[3] = {{}, {}, {}};
#pragma unroll
    for (int j = 0; j < 3; j++) {
      int t = wave + 8 * (g * 3 + j);         // t in [0,96)
      int rr = t - sel * 48;                  // tile within matrix
      mts[j] = rr & 3; nts[j] = rr >> 2;      // 4 M-tiles x 12 N-tiles
    }
    gemm_tiles<3>(As, CIN, wsel, CIN, 96, mts, nts, l16, half, acc);

    float* dst = (sel ? q : p) + (size_t)w * NODES * OEDGE;
#pragma unroll
    for (int j = 0; j < 3; j++) {
      int o = nts[j] * 16 + l16;
      float bias = sel ? 0.0f : bpq[o];
#pragma unroll
      for (int r = 0; r < 8; r++) {
        int node = mts[j] * 16 + r + half * 8;
        dst[node * OEDGE + o] = acc[j][r] + bias;
      }
    }
  }
}

// ---------------------------------------------------------------------------
// Kernel 4: h = max_k gelu(p[n] + q[idx[n,k]]) -> bf16 z (fc2 input)
// q window + idx staged via TDM; all 9 gathers are LDS reads.
// ---------------------------------------------------------------------------
__global__ __launch_bounds__(256) void edge_kernel(
    const float* __restrict__ p, const float* __restrict__ q,
    const int* __restrict__ idx, u16* __restrict__ z)
{
  __shared__ __align__(16) float qs[NODES * OEDGE];    // 48 KB
  __shared__ __align__(16) int   il[NODES * KNN];
  int w = blockIdx.x, tid = threadIdx.x;
  const float* qsrc = q + (size_t)w * NODES * OEDGE;
  const float* psrc = p + (size_t)w * NODES * OEDGE;
  __builtin_prefetch(psrc, 0, 1);                      // global_prefetch path

  if (tid < 32)      tdm_load_1d(qs, qsrc, NODES * OEDGE * 4);
  else if (tid < 64) tdm_load_1d(il, idx + (size_t)w * NODES * KNN,
                                 NODES * KNN * 4);
  tdm_wait_all();
  __syncthreads();

  u16* zdst = z + (size_t)w * NODES * OEDGE;
  for (int e = tid; e < NODES * OEDGE; e += 256) {
    int n = e / OEDGE, o = e - n * OEDGE;
    float pv = psrc[e];
    float mx = -3.0e38f;
#pragma unroll
    for (int k = 0; k < KNN; k++) {
      int m = il[n * KNN + k];
      float xv = pv + qs[m * OEDGE + o];
      float g = 0.5f * xv * (1.0f + erff(xv * 0.70710678118654752f)); // exact gelu
      mx = fmaxf(mx, g);
    }
    zdst[e] = f2bf(mx);
  }
}

// ---------------------------------------------------------------------------
// Kernel 5: fc2 (1x1 conv + affine) + residual, 64x192x96 GEMM per window,
// scatter back to NCHW. z and W2 staged via TDM.
// ---------------------------------------------------------------------------
__global__ __launch_bounds__(256) void fc2_kernel(
    const u16* __restrict__ z, const u16* __restrict__ W2,
    const float* __restrict__ b2, const float* __restrict__ x,
    float* __restrict__ out)
{
  __shared__ __align__(16) u16 Zs[NODES * OEDGE];   // 24.6 KB
  __shared__ __align__(16) u16 Ws[CIN * OEDGE];     // 36.9 KB
  int w = blockIdx.x, tid = threadIdx.x;
  int b = w >> 6, wy = (w >> 3) & 7, wx = w & 7;
  const u16* zsrc = z + (size_t)w * NODES * OEDGE;

  if (tid < 32)      tdm_load_1d(Zs, zsrc, NODES * OEDGE * 2);
  else if (tid < 64) tdm_load_1d(Ws, W2,   CIN * OEDGE * 2);
  tdm_wait_all();
  __syncthreads();

  int wave = tid >> 5, lane = tid & 31, half = lane >> 4, l16 = lane & 15;
  int mts[3], nts[3];
  f32x8 acc[3] = {{}, {}, {}};
#pragma unroll
  for (int j = 0; j < 3; j++) {                // 4 M-tiles x 6 N-tiles
    int t = wave + 8 * j; mts[j] = t & 3; nts[j] = t >> 2;
  }
  gemm_tiles<3>(Zs, OEDGE, Ws, OEDGE, OEDGE, mts, nts, l16, half, acc);

#pragma unroll
  for (int j = 0; j < 3; j++) {
    int oc = nts[j] * 16 + l16;
    float bias = b2[oc];
#pragma unroll
    for (int r = 0; r < 8; r++) {
      int node = mts[j] * 16 + r + half * 8;
      int gh = wy * 8 + (node >> 3), gw = wx * 8 + (node & 7);
      size_t gi = ((((size_t)b * 96 + oc) * 64 + gh) * 64 + gw);
      out[gi] = acc[j][r] + bias + x[gi];
    }
  }
}

// ---------------------------------------------------------------------------
extern "C" void kernel_launch(void* const* d_in, const int* in_sizes, int n_in,
                              void* d_out, int out_size, void* d_ws, size_t ws_size,
                              hipStream_t stream)
{
  (void)in_sizes; (void)n_in; (void)out_size; (void)ws_size;
  const float* x     = (const float*)d_in[0];
  const float* fc1_w = (const float*)d_in[1];
  const float* fc1_b = (const float*)d_in[2];
  const float* fc1_s = (const float*)d_in[3];
  const float* fc1_t = (const float*)d_in[4];
  const float* ew    = (const float*)d_in[5];
  const float* eb    = (const float*)d_in[6];
  const float* es    = (const float*)d_in[7];
  const float* et    = (const float*)d_in[8];
  const float* fc2_w = (const float*)d_in[9];
  const float* fc2_b = (const float*)d_in[10];
  const float* fc2_s = (const float*)d_in[11];
  const float* fc2_t = (const float*)d_in[12];

  char* base = (char*)d_ws;
  size_t off = 0;
  auto carve = [&](size_t bytes) -> void* {
    void* r = base + off;
    off = (off + bytes + 255) & ~(size_t)255;
    return r;
  };
  u16*   nodes_bf = (u16*)  carve((size_t)NWIN * NODES * CIN * 2);     // 12.6 MB
  int*   idxb     = (int*)  carve((size_t)NWIN * NODES * KNN * 4);     //  2.4 MB
  float* pbuf     = (float*)carve((size_t)NWIN * NODES * OEDGE * 4);   // 50.3 MB
  float* qbuf     = (float*)carve((size_t)NWIN * NODES * OEDGE * 4);   // 50.3 MB
  u16*   zbuf     = (u16*)  carve((size_t)NWIN * NODES * OEDGE * 2);   // 25.2 MB
  u16*   W1  = (u16*)  carve(9216 * 2);
  float* b1  = (float*)carve(96 * 4);
  u16*   Wp  = (u16*)  carve(18432 * 2);
  u16*   Wq  = (u16*)  carve(18432 * 2);
  float* bpq = (float*)carve(192 * 4);
  u16*   W2  = (u16*)  carve(18432 * 2);
  float* b2  = (float*)carve(96 * 4);

  prep_kernel<<<254, 256, 0, stream>>>(fc1_w, fc1_b, fc1_s, fc1_t,
                                       ew, eb, es, et,
                                       fc2_w, fc2_b, fc2_s, fc2_t,
                                       W1, b1, Wp, Wq, bpq, W2, b2);
  fc1_kernel <<<NWIN, 256, 0, stream>>>(x, W1, b1, nodes_bf);
  knn_kernel <<<NWIN, 256, 0, stream>>>(nodes_bf, idxb);
  pq_kernel  <<<NWIN, 256, 0, stream>>>(nodes_bf, Wp, Wq, bpq, pbuf, qbuf);
  edge_kernel<<<NWIN, 256, 0, stream>>>(pbuf, qbuf, idxb, zbuf);
  fc2_kernel <<<NWIN, 256, 0, stream>>>(zbuf, W2, b2, x, (float*)d_out);
}
